// FlashAttention_80169859548040
// MI455X (gfx1250) — matrix operand order repaired and checked by reading, not compiled, not run on hardware
//
#include <hip/hip_runtime.h>
#include <stdint.h>

#define DIM    1024
#define NHEADS 16
#define HDIM   64
#define BB     4
#define SS     2048
#define MTOK   (BB * SS)                  // 8192 tokens
#define ATT_SCALE2 0.18033688011112042f   // 64^-0.5 * log2(e)  (base-2 softmax)
#define SMALL  (-3.0e38f)                 // finite "masked" sentinel / penalty

typedef __attribute__((ext_vector_type(16))) _Float16 v16h;
typedef __attribute__((ext_vector_type(8)))  float    v8f;

union Frag { uint32_t u[8]; v16h h; };

__device__ __forceinline__ v8f wmma_f16(const Frag& a, const Frag& b, v8f c) {
  return __builtin_amdgcn_wmma_f32_16x16x32_f16(
      /*neg_a=*/false, a.h, /*neg_b=*/false, b.h,
      /*c_mod=*/(short)0, c, /*reuse_a=*/false, /*reuse_b=*/false);
}

// raw v_exp_f32 (no denormal-range guard; sentinel args underflow to 0)
__device__ __forceinline__ float fast_exp2(float x) {
  return __builtin_amdgcn_exp2f(x);
}

// A-matrix 16x32 f16 K-offset for packed half pair p (i = 2p), contiguous k = 16*hi + i (matches B)
__device__ __forceinline__ int a_koff(int p, int hi) {
  int i = 2 * p;
  return i + (hi << 4);
}

// ---------------------------------------------------------------- cast f32->f16
__global__ void cast_f32_to_f16(const float* __restrict__ src,
                                _Float16* __restrict__ dst, int n) {
  for (int i = blockIdx.x * blockDim.x + threadIdx.x; i < n;
       i += gridDim.x * blockDim.x)
    dst[i] = (_Float16)src[i];
}

// ---------------------------------------------------------------- GEMM: C = A * W^T + bias
// LDS-tiled: 128x128 block tile, k-step 32. A: [M,K] f16, W: [N,K] f16
// (B(k,n) = W[n,k] so W rows are contiguous). Each wave: 16-row strip x 128 cols.
// MODE 0: f16 store [M,N]; MODE 1: f16 transposed store Vt[(b*DIM+n)*SS + s];
// MODE 2: f32 store [M,N].
template <int MODE>
__global__ void gemm_wmma(const _Float16* __restrict__ A,
                          const _Float16* __restrict__ W,
                          const float* __restrict__ bias,
                          void* __restrict__ out, int M, int N, int K) {
  __shared__ __attribute__((aligned(16))) _Float16 la[128 * 32];
  __shared__ __attribute__((aligned(16))) _Float16 lb[128 * 32];

  const int tid = threadIdx.x;
  const int lane = tid & 31;
  const int wave = tid >> 5;
  const int hi = lane >> 4;
  const int ln = lane & 15;

  const int nblk = N >> 7;
  const int bm = blockIdx.x / nblk;
  const int bn = blockIdx.x % nblk;
  const int row0 = bm << 7;
  const int col0 = bn << 7;

  v8f acc[8];
#pragma unroll
  for (int c = 0; c < 8; ++c) acc[c] = (v8f){0.f,0.f,0.f,0.f,0.f,0.f,0.f,0.f};

  for (int kb = 0; kb < K; kb += 32) {
    __syncthreads();  // previous iteration's LDS reads complete
    // cooperative tile load: 512 16B-chunks per tile, 2 per thread per tile
#pragma unroll
    for (int i = 0; i < 2; ++i) {
      const int c0 = tid * 2 + i;            // 0..511
      const int r = c0 >> 2;                 // 0..127
      const int cc = (c0 & 3) << 3;          // half-offset 0,8,16,24
      *(uint4*)(la + r * 32 + cc) =
          *(const uint4*)(A + (size_t)(row0 + r) * K + kb + cc);
      *(uint4*)(lb + r * 32 + cc) =
          *(const uint4*)(W + (size_t)(col0 + r) * K + kb + cc);
    }
    __syncthreads();

    Frag a;
#pragma unroll
    for (int p = 0; p < 8; ++p)
      a.u[p] = *(const uint32_t*)(la + (wave * 16 + ln) * 32 + a_koff(p, hi));

#pragma unroll
    for (int c = 0; c < 8; ++c) {
      Frag b;
#pragma unroll
      for (int p = 0; p < 8; ++p)
        b.u[p] = *(const uint32_t*)(lb + (c * 16 + ln) * 32 + 2 * p + (hi << 4));
      acc[c] = wmma_f16(a, b, acc[c]);
    }
  }

#pragma unroll
  for (int c = 0; c < 8; ++c) {
    const int n = col0 + c * 16 + ln;
    const float bnv = bias[n];
#pragma unroll
    for (int r = 0; r < 8; ++r) {
      const int m = row0 + wave * 16 + r + (hi << 3);
      const float v = acc[c][r] + bnv;
      if (MODE == 0) {
        ((_Float16*)out)[(size_t)m * N + n] = (_Float16)v;
      } else if (MODE == 1) {
        const int bidx = m / SS, s = m - bidx * SS;  // N == DIM here
        ((_Float16*)out)[((size_t)bidx * DIM + n) * SS + s] = (_Float16)v;
      } else {
        ((float*)out)[(size_t)m * N + n] = v;
      }
    }
  }
}

// ---------------------------------------------------------------- attention
// One wave = one 16-row Q tile of one (b,h). Base-2 softmax, precomputed
// additive mask-penalty row in LDS, single-exponential online update,
// incremental pointers in the hot loops.
__global__ void attn_wmma(const _Float16* __restrict__ Q,
                          const _Float16* __restrict__ Km,
                          const _Float16* __restrict__ Vt,
                          const float* __restrict__ mask,
                          float* __restrict__ attn_out,   // (B,H,S,S) f32
                          _Float16* __restrict__ ctx) {   // (B,S,DIM) f16
  __shared__ _Float16 lds_p[8 * 16 * 32];   // per-wave P-tile staging (8 KB)
  __shared__ float    lds_pen[SS];          // mask penalty row (8 KB)

  const int tid = threadIdx.x;
  const int lane = tid & 31;
  const int wave = tid >> 5;
  const int hi = lane >> 4;
  const int ln = lane & 15;

  const int w = blockIdx.x * 8 + wave;
  const int qt = w & 127;          // S/16 = 128 tiles
  const int h  = (w >> 7) & 15;
  const int b  = w >> 11;
  const int q0 = qt << 4;

  // all 8 waves share batch b: stage penalty row once per block
  {
    const float* mrow = mask + (size_t)b * SS;
    for (int i = tid; i < SS; i += 256)
      lds_pen[i] = (mrow[i] == 0.f) ? SMALL : 0.f;
  }
  __syncthreads();

  // Q A-fragments (16x64 split into two 16x32 k-steps), loaded once
  const _Float16* qrow = Q + ((size_t)(b * SS + q0 + ln) * DIM + h * HDIM);
  Frag aq[2];
#pragma unroll
  for (int ks = 0; ks < 2; ++ks)
#pragma unroll
    for (int p = 0; p < 8; ++p)
      aq[ks].u[p] = *(const uint32_t*)(qrow + ks * 32 + a_koff(p, hi));

  float lm[8], ll[8];
#pragma unroll
  for (int r = 0; r < 8; ++r) { lm[r] = SMALL; ll[r] = 0.f; }

  // ---------------- pass 1: lane-local online softmax stats (base-2) --------
  const _Float16* kptr = Km + ((size_t)(b * SS + ln) * DIM + h * HDIM);
  for (int j = 0; j < SS / 16; ++j, kptr += 16 * DIM) {
    v8f s = {0.f, 0.f, 0.f, 0.f, 0.f, 0.f, 0.f, 0.f};
#pragma unroll
    for (int ks = 0; ks < 2; ++ks) {
      Frag bk;
#pragma unroll
      for (int p = 0; p < 8; ++p)
        bk.u[p] = *(const uint32_t*)(kptr + ks * 32 + 2 * p + (hi << 4));
      s = wmma_f16(aq[ks], bk, s);
    }
    const float pen = lds_pen[(j << 4) + ln];
#pragma unroll
    for (int r = 0; r < 8; ++r) {
      const float sv = __builtin_fmaf(s[r], ATT_SCALE2, pen);
      // single-exponential online update: one of the two exps is always 2^0.
      const float d = sv - lm[r];
      const bool keep = (d <= 0.f);
      const float e = fast_exp2(-__builtin_fabsf(d));
      const float lla = ll[r] + e;                      // max unchanged
      const float llb = __builtin_fmaf(ll[r], e, 1.f);  // new max = sv
      ll[r] = keep ? lla : llb;
      lm[r] = keep ? lm[r] : sv;
    }
  }

  // cross-lane merge (once): M = max_i m_i ; L = sum_i l_i * 2^(m_i - M)
  float rowm[8], invl[8];
#pragma unroll
  for (int r = 0; r < 8; ++r) {
    float Mx = lm[r];
    Mx = fmaxf(Mx, __shfl_xor(Mx, 1, 32));
    Mx = fmaxf(Mx, __shfl_xor(Mx, 2, 32));
    Mx = fmaxf(Mx, __shfl_xor(Mx, 4, 32));
    Mx = fmaxf(Mx, __shfl_xor(Mx, 8, 32));
    float t = ll[r] * fast_exp2(lm[r] - Mx);
    t += __shfl_xor(t, 1, 32);
    t += __shfl_xor(t, 2, 32);
    t += __shfl_xor(t, 4, 32);
    t += __shfl_xor(t, 8, 32);
    rowm[r] = Mx;
    invl[r] = (t > 0.f) ? 1.f / t : 0.f;
  }

  // ---------------- pass 2: write probs + P*V ----------------
  v8f oacc[4];
#pragma unroll
  for (int c = 0; c < 4; ++c) oacc[c] = (v8f){0.f,0.f,0.f,0.f,0.f,0.f,0.f,0.f};

  _Float16* myp = lds_p + wave * 512;

  const _Float16* kptr2 = Km + ((size_t)(b * SS + ln) * DIM + h * HDIM);
  const _Float16* vptr  = Vt + ((size_t)(b * DIM + h * HDIM + ln) * SS);
  float* aptr = attn_out + ((size_t)(b * NHEADS + h) * SS + q0) * SS + ln;

  for (int j2 = 0; j2 < SS / 32;
       ++j2, kptr2 += 32 * DIM, vptr += 32, aptr += 32) {
#pragma unroll
    for (int t = 0; t < 2; ++t) {
      const _Float16* krow = kptr2 + t * (16 * DIM);
      v8f s = {0.f, 0.f, 0.f, 0.f, 0.f, 0.f, 0.f, 0.f};
#pragma unroll
      for (int ks = 0; ks < 2; ++ks) {
        Frag bk;
#pragma unroll
        for (int p = 0; p < 8; ++p)
          bk.u[p] = *(const uint32_t*)(krow + ks * 32 + 2 * p + (hi << 4));
        s = wmma_f16(aq[ks], bk, s);
      }
      const float pen = lds_pen[(j2 << 5) + (t << 4) + ln];
#pragma unroll
      for (int r = 0; r < 8; ++r) {
        const float sv = __builtin_fmaf(s[r], ATT_SCALE2, pen);
        const float pr = fast_exp2(sv - rowm[r]) * invl[r];
        const int m = r + (hi << 3);
        __builtin_nontemporal_store(pr, aptr + (size_t)m * SS + (t << 4));
        myp[m * 32 + (t << 4) + ln] = (_Float16)pr;
      }
    }
    // no block barrier: myp slice is wave-private; same-wave DS ops are
    // in-order and the compiler inserts the required s_wait_dscnt.

    Frag ap;
#pragma unroll
    for (int p = 0; p < 8; ++p)
      ap.u[p] = *(const uint32_t*)(myp + ln * 32 + a_koff(p, hi));

#pragma unroll
    for (int c = 0; c < 4; ++c) {
      const _Float16* vrow = vptr + (size_t)(c * 16) * SS;
      Frag bv;
#pragma unroll
      for (int p = 0; p < 8; ++p)
        bv.u[p] = *(const uint32_t*)(vrow + 2 * p + (hi << 4));
      oacc[c] = wmma_f16(ap, bv, oacc[c]);
    }
  }

#pragma unroll
  for (int c = 0; c < 4; ++c)
#pragma unroll
    for (int r = 0; r < 8; ++r) {
      const int m = q0 + r + (hi << 3);
      ctx[((size_t)b * SS + m) * DIM + h * HDIM + c * 16 + ln] =
          (_Float16)oacc[c][r];
    }
}

// ---------------------------------------------------------------- launch
extern "C" void kernel_launch(void* const* d_in, const int* in_sizes, int n_in,
                              void* d_out, int out_size, void* d_ws,
                              size_t ws_size, hipStream_t stream) {
  const float* x    = (const float*)d_in[0];
  const float* mask = (const float*)d_in[1];
  const float* Wq   = (const float*)d_in[2];
  const float* bq   = (const float*)d_in[3];
  const float* Wk   = (const float*)d_in[4];
  const float* bk   = (const float*)d_in[5];
  const float* Wv   = (const float*)d_in[6];
  const float* bv   = (const float*)d_in[7];
  const float* Wo   = (const float*)d_in[8];
  const float* bo   = (const float*)d_in[9];
  float* out = (float*)d_out;

  // workspace carve-up (f16 elements)
  _Float16* x16  = (_Float16*)d_ws;
  _Float16* Wq16 = x16  + (size_t)MTOK * DIM;
  _Float16* Wk16 = Wq16 + (size_t)DIM * DIM;
  _Float16* Wv16 = Wk16 + (size_t)DIM * DIM;
  _Float16* Wo16 = Wv16 + (size_t)DIM * DIM;
  _Float16* Q16  = Wo16 + (size_t)DIM * DIM;
  _Float16* K16  = Q16  + (size_t)MTOK * DIM;
  _Float16* Vt16 = K16  + (size_t)MTOK * DIM;
  _Float16* C16  = Vt16 + (size_t)MTOK * DIM;

  cast_f32_to_f16<<<1024, 256, 0, stream>>>(x, x16, MTOK * DIM);
  cast_f32_to_f16<<<512, 256, 0, stream>>>(Wq, Wq16, DIM * DIM);
  cast_f32_to_f16<<<512, 256, 0, stream>>>(Wk, Wk16, DIM * DIM);
  cast_f32_to_f16<<<512, 256, 0, stream>>>(Wv, Wv16, DIM * DIM);
  cast_f32_to_f16<<<512, 256, 0, stream>>>(Wo, Wo16, DIM * DIM);

  const int gemm_blocks = (MTOK / 128) * (DIM / 128);  // 64*8 = 512
  gemm_wmma<0><<<gemm_blocks, 256, 0, stream>>>(x16, Wq16, bq, Q16, MTOK, DIM, DIM);
  gemm_wmma<0><<<gemm_blocks, 256, 0, stream>>>(x16, Wk16, bk, K16, MTOK, DIM, DIM);
  gemm_wmma<1><<<gemm_blocks, 256, 0, stream>>>(x16, Wv16, bv, Vt16, MTOK, DIM, DIM);

  float* attn_out = out + (size_t)MTOK * DIM;  // second tuple element
  attn_wmma<<<(BB * NHEADS * (SS / 16)) / 8, 256, 0, stream>>>(
      Q16, K16, Vt16, mask, attn_out, C16);

  gemm_wmma<2><<<gemm_blocks, 256, 0, stream>>>(C16, Wo16, bo, out, MTOK, DIM, DIM);
}
